// MemA2M_40407052321014
// MI455X (gfx1250) — matrix operand order repaired and checked by reading, not compiled, not run on hardware
//
#include <hip/hip_runtime.h>
#include <hip/hip_bf16.h>

// ---------------------------------------------------------------------------
// MI455X / gfx1250: every GEMM-shaped op (conv-as-GEMM, QKV, attention, proj,
// MLP) runs on v_wmma_f32_16x16x32_bf16; LN / softmax / BN epilogues / the
// rank-8 memory attention stay f32 VALU.  Wave32; attention blocks are 64
// threads (2 waves) so K^T + P + rpb fit in static LDS; GEMM blocks are 128.
// ---------------------------------------------------------------------------

typedef __attribute__((ext_vector_type(16))) __bf16 v16bf;
typedef __attribute__((ext_vector_type(8)))  __bf16 v8bf;
typedef __attribute__((ext_vector_type(8)))  float  v8f;
typedef v16bf v16bf_a16 __attribute__((aligned(16)));   // 16B-aligned 32B vector load

#define DEV static __device__ __forceinline__

DEV __bf16 f2bf(float f) {                       // round-to-nearest-even f32->bf16
  unsigned u = __builtin_bit_cast(unsigned, f);
  unsigned r = u + 0x7FFFu + ((u >> 16) & 1u);
  unsigned short h = (unsigned short)(r >> 16);
  return __builtin_bit_cast(__bf16, h);
}

DEV v16bf cat16(v8bf lo, v8bf hi) {
  return __builtin_shufflevector(lo, hi, 0,1,2,3,4,5,6,7,8,9,10,11,12,13,14,15);
}

DEV v8f wmma_bf16(v16bf a, v16bf b, v8f c) {
  // 8 args: (neg_a, A, neg_b, B, c_mod, C, reuse_a, reuse_b)
  return __builtin_amdgcn_wmma_f32_16x16x32_bf16(false, a, false, b, (short)0, c,
                                                 false, false);
}

DEV float gelu_f(float v) {                      // tanh approximation (jax default)
  float u = 0.7978845608028654f * (v + 0.044715f * v * v * v);
  return 0.5f * v * (1.0f + tanhf(u));
}

DEV int regid(int v) { return (v < 48) ? 0 : ((v < 56) ? 1 : 2); }  // shift=8 mask regions

// ---------------------------------------------------------------------------
// BN folding: sc = gamma*rsqrt(var+eps); sb = beta - mean*sc  (256 channels)
// ---------------------------------------------------------------------------
__global__ __launch_bounds__(256)
void bn_prep(const float* __restrict__ g, const float* __restrict__ be,
             const float* __restrict__ mean, const float* __restrict__ var,
             float* __restrict__ sc, float* __restrict__ sb) {
  const int c = threadIdx.x;
  const float s = g[c] * rsqrtf(var[c] + 1e-5f);
  sc[c] = s;
  sb[c] = be[c] - mean[c] * s;
}

// ---------------------------------------------------------------------------
// im2col for the 3x3 conv (64 in-ch, zero pad) -> bf16 [32768, 576] rows.
// One pixel per block; k index = ci*9 + (ky*3+kx) matching conv_w layout.
// ---------------------------------------------------------------------------
__global__ __launch_bounds__(256)
void im2col(const float* __restrict__ x, __bf16* __restrict__ out) {
  const int pix = blockIdx.x;                    // b*4096 + y*64 + x
  const int b = pix >> 12, y = (pix >> 6) & 63, xx = pix & 63;
  __bf16* dst = out + (size_t)pix * 576;
  for (int k = threadIdx.x; k < 576; k += 256) {
    const int ci = k / 9, p = k % 9;
    const int yy = y + p / 3 - 1, xs = xx + p % 3 - 1;
    float v = 0.f;
    if (yy >= 0 && yy < 64 && xs >= 0 && xs < 64)
      v = x[(((size_t)(b * 64 + ci)) * 64 + yy) * 64 + xs];
    dst[k] = f2bf(v);
  }
}

// ---------------------------------------------------------------------------
// Convert f32 weight [N,K] -> bf16 [N,K], no transpose: WMMA B-fragments become
// two 16-byte vector loads (lane = N row, 2x8 contiguous K per lane).
// ---------------------------------------------------------------------------
__global__ __launch_bounds__(256)
void cvt_wT(const float* __restrict__ w, __bf16* __restrict__ wt, int N, int K) {
  const int idx = blockIdx.x * 256 + threadIdx.x;
  if (idx >= N * K) return;
  const int n = idx / K, k = idx % K;
  (void)n; (void)k; wt[idx] = f2bf(w[idx]);
}

// ---------------------------------------------------------------------------
// LN1 + cyclic shift + window partition -> bf16 windowed tokens.
// One wave per destination token; shfl_xor reductions for mean/var.
// ---------------------------------------------------------------------------
__global__ __launch_bounds__(128)
void ln_window(const float* __restrict__ t, const float* __restrict__ g,
               const float* __restrict__ b, __bf16* __restrict__ out, int shift) {
  const int tok = blockIdx.x * 4 + (threadIdx.x >> 5);
  const int lane = threadIdx.x & 31;
  const int win = tok >> 8, n = tok & 255;
  const int bi = win >> 4, wy = (win >> 2) & 3, wx = win & 3;
  const int sy = (wy * 16 + (n >> 4) + shift) & 63;
  const int sx = (wx * 16 + (n & 15) + shift) & 63;
  const float* src = t + ((size_t)bi * 4096 + sy * 64 + sx) * 256;
  float v[8]; float s = 0.f;
#pragma unroll
  for (int j = 0; j < 8; ++j) { v[j] = src[lane + j * 32]; s += v[j]; }
#pragma unroll
  for (int off = 16; off >= 1; off >>= 1) s += __shfl_xor(s, off, 32);
  const float mu = s * (1.f / 256.f);
  float q = 0.f;
#pragma unroll
  for (int j = 0; j < 8; ++j) { const float d = v[j] - mu; q += d * d; }
#pragma unroll
  for (int off = 16; off >= 1; off >>= 1) q += __shfl_xor(q, off, 32);
  const float inv = rsqrtf(q * (1.f / 256.f) + 1e-5f);
  __bf16* dst = out + (size_t)tok * 256;
#pragma unroll
  for (int j = 0; j < 8; ++j) {
    const int c = lane + j * 32;
    dst[c] = f2bf((v[j] - mu) * inv * g[c] + b[c]);
  }
}

// ---------------------------------------------------------------------------
// WMMA GEMM: C[M,N] = A[M,K](bf16 row-major) * Wt[N,K]^T(bf16 row-major) + epilogue.
// 128 threads = 4 waves; each wave owns a 16x128 strip (8 accumulator tiles).
// EPI: 0 bias->bf16 | 1 bias->f32 | 2 bias+GELU->bf16
//      3 bias+residual(aux[m*N+c])->f32 | 4 acc*bias[c]+aux[c],ReLU->f32 (BN)
// ---------------------------------------------------------------------------
template <int EPI>
__global__ __launch_bounds__(128)
void gemm_bf16(const __bf16* __restrict__ A, const __bf16* __restrict__ Wt,
               const float* __restrict__ bias, void* __restrict__ outv,
               const float* __restrict__ aux, int M, int N, int K) {
  const int wave = threadIdx.x >> 5, lane = threadIdx.x & 31;
  const int row0 = blockIdx.x * 64 + wave * 16;
  const int col0 = blockIdx.y * 128;
  const int mrow = row0 + (lane & 15);
  const int kofs = (lane < 16) ? 0 : 8;          // ISA A-fragment lane->K mapping
  v8f acc[8];
#pragma unroll
  for (int tt = 0; tt < 8; ++tt) acc[tt] = (v8f){0, 0, 0, 0, 0, 0, 0, 0};
  const __bf16* arow = A + (size_t)mrow * K + kofs;
  const int nk = K >> 5;
  for (int kk = 0; kk < nk; ++kk) {
    const v16bf af = cat16(*(const v8bf*)(arow + kk * 32),
                           *(const v8bf*)(arow + kk * 32 + 16));
    const __bf16* wrow = Wt + (size_t)(col0 + (lane & 15)) * K + kk * 32 + kofs;
#pragma unroll
    for (int tt = 0; tt < 8; ++tt)
      acc[tt] = wmma_bf16(af, cat16(*(const v8bf*)(wrow + (size_t)tt * 16 * K), *(const v8bf*)(wrow + (size_t)tt * 16 * K + 16)), acc[tt]);
  }
  const int colc = lane & 15, radd = (lane < 16) ? 0 : 8;
#pragma unroll
  for (int tt = 0; tt < 8; ++tt)
#pragma unroll
    for (int r = 0; r < 8; ++r) {
      const int m = row0 + r + radd;
      const int c = col0 + tt * 16 + colc;
      float v;
      if (EPI == 4) v = fmaxf(acc[tt][r] * bias[c] + aux[c], 0.f);
      else          v = acc[tt][r] + bias[c];
      if (EPI == 2) v = gelu_f(v);
      if (EPI == 3) v += aux[(size_t)m * N + c];
      if (EPI == 0 || EPI == 2)
        ((__bf16*)outv)[(size_t)m * N + c] = f2bf(v);
      else
        ((float*)outv)[(size_t)m * N + c] = v;
    }
}

// ---------------------------------------------------------------------------
// Windowed multi-head attention. One 64-thread block (2 waves) per
// (window, head).  K^T is transposed into LDS once (unused; B-frags are loaded
// from global K rows), then each wave runs 8 blocks of 16 query rows:
// Q*K^T (WMMA) -> scale + rel-pos bias + shift mask -> rowwise softmax
// (shfl_xor within 16-lane halves) -> P to LDS -> P*V (WMMA) -> o (f32).
// ---------------------------------------------------------------------------
#define KT_STRIDE 264
__global__ __launch_bounds__(64)
void win_attn(const __bf16* __restrict__ qkv, const float* __restrict__ rpb,
              float* __restrict__ o, int shift) {
  __shared__ __align__(32) __bf16 KtS[64 * KT_STRIDE];   // 33 KB  K^T [feat][key]
  __shared__ __align__(32) __bf16 P_s[2][16 * 256];      // 16 KB  per-wave P
  __shared__ __align__(16) float  rpb_s[961];            //  4 KB
  const int win = blockIdx.x, head = blockIdx.y;
  const int wave = threadIdx.x >> 5, lane = threadIdx.x & 31;
  const size_t base = (size_t)win * 256 * 768;
  const __bf16* qb = qkv + base + head * 64;
  const __bf16* kb = qkv + base + 256 + head * 64;
  const __bf16* vb = qkv + base + 512 + head * 64;
  for (int idx = threadIdx.x; idx < 961; idx += 64) rpb_s[idx] = rpb[idx * 4 + head];
  // stage K^T: coalesced global reads (64 contiguous feats per key row)
  for (int idx = threadIdx.x; idx < 64 * 256; idx += 64) {
    const int key = idx >> 6, f = idx & 63;
    KtS[f * KT_STRIDE + key] = kb[(size_t)key * 768 + f];
  }
  __syncthreads();
  const int wy = (win >> 2) & 3, wx = win & 3;   // window position inside image
  const int colc = lane & 15, radd = (lane < 16) ? 0 : 8;
  const int kofs = (lane < 16) ? 0 : 8;
  __bf16* pw = &P_s[wave][0];

  for (int rb = 0; rb < 8; ++rb) {
    const int row0 = wave * 128 + rb * 16;
    const int mrow = row0 + (lane & 15);
    const __bf16* qr = qb + (size_t)mrow * 768 + kofs;
    v16bf aq[2];
#pragma unroll
    for (int kk = 0; kk < 2; ++kk)
      aq[kk] = cat16(*(const v8bf*)(qr + kk * 32), *(const v8bf*)(qr + kk * 32 + 16));

    // ---- scores: S[16 col tiles] = Q * K^T, B-frags straight from global K rows ----
    v8f S[16];
#pragma unroll
    for (int tt = 0; tt < 16; ++tt) {
      v8f s = (v8f){0, 0, 0, 0, 0, 0, 0, 0};
#pragma unroll
      for (int kk = 0; kk < 2; ++kk) {
        const __bf16* kr = kb + (size_t)(tt * 16 + colc) * 768 + kk * 32 + kofs;
        s = wmma_bf16(aq[kk],
                      cat16(*(const v8bf*)kr, *(const v8bf*)(kr + 16)), s);
      }
      S[tt] = s;
    }

    // ---- scale + rel-pos bias + shift mask + rowwise softmax ----
#pragma unroll
    for (int r = 0; r < 8; ++r) {
      const int m = row0 + r + radd;
      const int my = m >> 4, mx2 = m & 15;
      int rm = 0;
      if (shift) rm = regid(wy * 16 + my) * 3 + regid(wx * 16 + mx2);
      float rmax = -1e30f;
#pragma unroll
      for (int tt = 0; tt < 16; ++tt) {
        const int n = tt * 16 + colc;
        const int ny = n >> 4, nx = n & 15;
        float v = S[tt][r] * 0.125f + rpb_s[(my - ny + 15) * 31 + (mx2 - nx + 15)];
        if (shift) {
          const int rn = regid(wy * 16 + ny) * 3 + regid(wx * 16 + nx);
          if (rn != rm) v -= 100.0f;
        }
        S[tt][r] = v;
        rmax = fmaxf(rmax, v);
      }
#pragma unroll
      for (int off = 8; off >= 1; off >>= 1)        // stays inside 16-lane half
        rmax = fmaxf(rmax, __shfl_xor(rmax, off, 32));
      float sum = 0.f;
#pragma unroll
      for (int tt = 0; tt < 16; ++tt) {
        const float e = __expf(S[tt][r] - rmax);
        S[tt][r] = e; sum += e;
      }
#pragma unroll
      for (int off = 8; off >= 1; off >>= 1) sum += __shfl_xor(sum, off, 32);
      const float inv = 1.0f / sum;
      const int prow = r + radd;
#pragma unroll
      for (int tt = 0; tt < 16; ++tt)
        pw[prow * 256 + tt * 16 + colc] = f2bf(S[tt][r] * inv);
    }
    __syncthreads();

    // ---- O = P * V: A from LDS P, B = per-element strided gathers down V columns ----
    v8f O[4];
#pragma unroll
    for (int tt = 0; tt < 4; ++tt) O[tt] = (v8f){0, 0, 0, 0, 0, 0, 0, 0};
    const __bf16* pr = pw + (lane & 15) * 256 + kofs;
    for (int kk = 0; kk < 8; ++kk) {
      const v16bf ap = cat16(*(const v8bf*)(pr + kk * 32),
                             *(const v8bf*)(pr + kk * 32 + 16));
      const __bf16* vr = vb + (size_t)(kk * 32 + kofs) * 768 + colc;
#pragma unroll
      for (int tt = 0; tt < 4; ++tt) {
        v16bf bv;
#pragma unroll
        for (int s = 0; s < 16; ++s)
          bv[s] = vr[(size_t)(s + (s >= 8 ? 8 : 0)) * 768 + tt * 16];
        O[tt] = wmma_bf16(ap, bv, O[tt]);
      }
    }
#pragma unroll
    for (int tt = 0; tt < 4; ++tt)
#pragma unroll
      for (int r = 0; r < 8; ++r) {
        const int m = row0 + r + radd;
        const int c = head * 64 + tt * 16 + colc;
        o[((size_t)win * 256 + m) * 256 + c] = O[tt][r];
      }
    __syncthreads();
  }
}

// ---------------------------------------------------------------------------
// Low-rank memory cross-attention (rank 8, 128 slots).  One wave per token.
// o2 = bf16(o + 0.1 * up(softmax(down(o) . mem^T) @ mem))
// ---------------------------------------------------------------------------
__global__ __launch_bounds__(128)
void mem_attn(const float* __restrict__ o, const float* __restrict__ dw,
              const float* __restrict__ db, const float* __restrict__ mb,
              const float* __restrict__ uw, const float* __restrict__ ub,
              __bf16* __restrict__ o2) {
  const int tok = blockIdx.x * 4 + (threadIdx.x >> 5);
  const int lane = threadIdx.x & 31;
  const float* orow = o + (size_t)tok * 256;
  float ov[8];
#pragma unroll
  for (int j = 0; j < 8; ++j) ov[j] = orow[lane + j * 32];
  float low[8];
#pragma unroll
  for (int e = 0; e < 8; ++e) {
    float s = 0.f;
#pragma unroll
    for (int j = 0; j < 8; ++j) s += ov[j] * dw[e * 256 + lane + j * 32];
#pragma unroll
    for (int off = 16; off >= 1; off >>= 1) s += __shfl_xor(s, off, 32);
    low[e] = s + db[e];
  }
  float sc[4]; float mx = -1e30f;
#pragma unroll
  for (int p = 0; p < 4; ++p) {
    const int slot = lane + p * 32;
    float s = 0.f;
#pragma unroll
    for (int e = 0; e < 8; ++e) s += low[e] * mb[slot * 8 + e];
    sc[p] = s; mx = fmaxf(mx, s);
  }
#pragma unroll
  for (int off = 16; off >= 1; off >>= 1) mx = fmaxf(mx, __shfl_xor(mx, off, 32));
  float sum = 0.f;
#pragma unroll
  for (int p = 0; p < 4; ++p) { sc[p] = __expf(sc[p] - mx); sum += sc[p]; }
#pragma unroll
  for (int off = 16; off >= 1; off >>= 1) sum += __shfl_xor(sum, off, 32);
  const float inv = 1.f / sum;
  float agg[8];
#pragma unroll
  for (int e = 0; e < 8; ++e) agg[e] = 0.f;
#pragma unroll
  for (int p = 0; p < 4; ++p) {
    const int slot = lane + p * 32;
    const float a = sc[p] * inv;
#pragma unroll
    for (int e = 0; e < 8; ++e) agg[e] += a * mb[slot * 8 + e];
  }
#pragma unroll
  for (int e = 0; e < 8; ++e)
#pragma unroll
    for (int off = 16; off >= 1; off >>= 1) agg[e] += __shfl_xor(agg[e], off, 32);
  __bf16* dst = o2 + (size_t)tok * 256;
#pragma unroll
  for (int j = 0; j < 8; ++j) {
    const int c = lane + j * 32;
    float u = ub[c];
#pragma unroll
    for (int e = 0; e < 8; ++e) u += agg[e] * uw[c * 8 + e];
    dst[c] = f2bf(ov[j] + 0.1f * u);
  }
}

// ---------------------------------------------------------------------------
// Window reverse + un-shift + residual add into t, then LN2 -> h (bf16).
// ---------------------------------------------------------------------------
__global__ __launch_bounds__(128)
void unwin_add_ln(float* __restrict__ t, const float* __restrict__ z,
                  const float* __restrict__ g, const float* __restrict__ b,
                  __bf16* __restrict__ h, int shift) {
  const int tok = blockIdx.x * 4 + (threadIdx.x >> 5);
  const int lane = threadIdx.x & 31;
  const int bi = tok >> 12, y = (tok >> 6) & 63, x = tok & 63;
  const int yy = (y - shift) & 63, xx = (x - shift) & 63;
  const int win = bi * 16 + (yy >> 4) * 4 + (xx >> 4);
  const int n = (yy & 15) * 16 + (xx & 15);
  const float* zr = z + ((size_t)win * 256 + n) * 256;
  float* tr = t + (size_t)tok * 256;
  float v[8]; float s = 0.f;
#pragma unroll
  for (int j = 0; j < 8; ++j) {
    const int c = lane + j * 32;
    v[j] = tr[c] + zr[c];
    tr[c] = v[j];
    s += v[j];
  }
#pragma unroll
  for (int off = 16; off >= 1; off >>= 1) s += __shfl_xor(s, off, 32);
  const float mu = s * (1.f / 256.f);
  float q = 0.f;
#pragma unroll
  for (int j = 0; j < 8; ++j) { const float d = v[j] - mu; q += d * d; }
#pragma unroll
  for (int off = 16; off >= 1; off >>= 1) q += __shfl_xor(q, off, 32);
  const float inv = rsqrtf(q * (1.f / 256.f) + 1e-5f);
  __bf16* dst = h + (size_t)tok * 256;
#pragma unroll
  for (int j = 0; j < 8; ++j) {
    const int c = lane + j * 32;
    dst[c] = f2bf((v[j] - mu) * inv * g[c] + b[c]);
  }
}

// ---------------------------------------------------------------------------
// Final NHWC -> NCHW.
// ---------------------------------------------------------------------------
__global__ __launch_bounds__(256)
void nhwc2nchw(const float* __restrict__ t, float* __restrict__ out) {
  const int idx = blockIdx.x * 256 + threadIdx.x;   // ((b*256+c)*64+y)*64+x
  const int b = idx >> 20, c = (idx >> 12) & 255;
  const int y = (idx >> 6) & 63, x = idx & 63;
  out[idx] = t[((size_t)b * 4096 + y * 64 + x) * 256 + c];
}

// ---------------------------------------------------------------------------
extern "C" void kernel_launch(void* const* d_in, const int* in_sizes, int n_in,
                              void* d_out, int out_size, void* d_ws, size_t ws_size,
                              hipStream_t stream) {
  (void)in_sizes; (void)n_in; (void)out_size; (void)ws_size;
  const float* x       = (const float*)d_in[0];
  const float* conv_w  = (const float*)d_in[1];
  const float* bng     = (const float*)d_in[2];
  const float* bnb     = (const float*)d_in[3];
  const float* bnm     = (const float*)d_in[4];
  const float* bnv     = (const float*)d_in[5];
  const float* ln1_g   = (const float*)d_in[6];
  const float* ln1_b   = (const float*)d_in[7];
  const float* qkv_w   = (const float*)d_in[8];
  const float* qkv_b   = (const float*)d_in[9];
  const float* proj_w  = (const float*)d_in[10];
  const float* proj_b  = (const float*)d_in[11];
  const float* rpb     = (const float*)d_in[12];
  const float* memb    = (const float*)d_in[13];
  const float* down_w  = (const float*)d_in[14];
  const float* down_b  = (const float*)d_in[15];
  const float* up_w    = (const float*)d_in[16];
  const float* up_b    = (const float*)d_in[17];
  const float* ln2_g   = (const float*)d_in[18];
  const float* ln2_b   = (const float*)d_in[19];
  const float* mlp_w1  = (const float*)d_in[20];
  const float* mlp_b1  = (const float*)d_in[21];
  const float* mlp_w2  = (const float*)d_in[22];
  const float* mlp_b2  = (const float*)d_in[23];

  // ---- workspace layout (bytes) ----
  char* ws = (char*)d_ws;
  float*  t    = (float*)(ws);                       // 32 MB  [32768,256] f32, persistent
  __bf16* winX = (__bf16*)(ws + 33554432ull);        // 16 MB  LN1 windowed; reused as h
  __bf16* qkvb = (__bf16*)(ws + 50331648ull);        // 48 MB  im2col -> qkv -> mlp1
  float*  obuf = (float*)(ws + 100663296ull);        // 32 MB  attn o; reused as proj z
  __bf16* o2   = (__bf16*)(ws + 134217728ull);       // 16 MB  mem-attn out (bf16)
  __bf16* wb   = (__bf16*)(ws + 150994944ull);       //  ~4.5 MB bf16 weights, [N,K] as given
  __bf16* qkvT  = wb;                                // 4 * 196608
  __bf16* projT = qkvT + 786432;                     // 4 * 65536
  __bf16* w1T   = projT + 262144;                    // 4 * 131072
  __bf16* w2T   = w1T + 524288;                      // 4 * 131072
  __bf16* convT = w2T + 524288;                      // 576 * 256
  float*  bnsc  = (float*)(ws + 150994944ull + 4489216ull);   // 256 f32
  float*  bnsb  = bnsc + 256;                                  // 256 f32

  // ---- stem: conv3x3+BN+ReLU as im2col + WMMA GEMM with fused BN epilogue --
  bn_prep<<<1, 256, 0, stream>>>(bng, bnb, bnm, bnv, bnsc, bnsb);
  cvt_wT<<<576, 256, 0, stream>>>(conv_w, convT, 256, 576);
  im2col<<<32768, 256, 0, stream>>>(x, qkvb);
  gemm_bf16<4><<<dim3(512, 2), 128, 0, stream>>>(qkvb, convT, bnsc, t, bnsb,
                                                 32768, 256, 576);

  // ---- transformer weights: convert to bf16, layout kept (W[N,K] -> Wt[N,K]) --
  for (int i = 0; i < 4; ++i) {
    cvt_wT<<<768, 256, 0, stream>>>(qkv_w  + (size_t)i * 196608, qkvT  + (size_t)i * 196608, 768, 256);
    cvt_wT<<<256, 256, 0, stream>>>(proj_w + (size_t)i * 65536,  projT + (size_t)i * 65536,  256, 256);
    cvt_wT<<<512, 256, 0, stream>>>(mlp_w1 + (size_t)i * 131072, w1T   + (size_t)i * 131072, 512, 256);
    cvt_wT<<<512, 256, 0, stream>>>(mlp_w2 + (size_t)i * 131072, w2T   + (size_t)i * 131072, 256, 512);
  }

  const int shifts[4] = {0, 8, 0, 8};
  for (int i = 0; i < 4; ++i) {
    const int sh = shifts[i];
    ln_window<<<8192, 128, 0, stream>>>(t, ln1_g + i * 256, ln1_b + i * 256, winX, sh);
    gemm_bf16<0><<<dim3(512, 6), 128, 0, stream>>>(winX, qkvT + (size_t)i * 196608,
        qkv_b + i * 768, qkvb, nullptr, 32768, 768, 256);
    win_attn<<<dim3(128, 4), 64, 0, stream>>>(qkvb, rpb + (size_t)i * 3844, obuf, sh);
    mem_attn<<<8192, 128, 0, stream>>>(obuf, down_w + i * 2048, down_b + i * 8,
        memb + i * 1024, up_w + i * 2048, up_b + i * 256, o2);
    gemm_bf16<1><<<dim3(512, 2), 128, 0, stream>>>(o2, projT + (size_t)i * 65536,
        proj_b + i * 256, obuf, nullptr, 32768, 256, 256);
    unwin_add_ln<<<8192, 128, 0, stream>>>(t, obuf, ln2_g + i * 256, ln2_b + i * 256,
        winX, sh);
    gemm_bf16<2><<<dim3(512, 4), 128, 0, stream>>>(winX, w1T + (size_t)i * 131072,
        mlp_b1 + i * 512, qkvb, nullptr, 32768, 512, 256);
    gemm_bf16<3><<<dim3(512, 2), 128, 0, stream>>>(qkvb, w2T + (size_t)i * 131072,
        mlp_b2 + i * 256, t, t, 32768, 256, 512);
  }

  nhwc2nchw<<<32768, 256, 0, stream>>>(t, (float*)d_out);
}
